// FrameTransformerDecoder_14362370638500
// MI455X (gfx1250) — matrix operand order repaired and checked by reading, not compiled, not run on hardware
//
#include <hip/hip_runtime.h>
#include <hip/hip_bf16.h>
#include <cstdint>
#include <cstddef>

// ---------------------------------------------------------------------------
// Types / constants
// ---------------------------------------------------------------------------
typedef __bf16 bfloat;
typedef bfloat v16bf __attribute__((ext_vector_type(16)));
typedef float  v8f   __attribute__((ext_vector_type(8)));

namespace {
constexpr int  kB = 4, kC = 8, kBINS = 1024, kW = 512, kNB = 4, kD = 256, kFF = 2048;
constexpr long kRows = (long)kB * kW;              // 2048 token rows
constexpr long kOut0 = (long)kB * kBINS * kW;      // 2,097,152
constexpr long kAtt  = (long)kB * kNB * kW * kW;   // 4,194,304
constexpr float kEps = 1e-5f;
}

union FragBf { v16bf v; uint4 q[2]; };

// ---- CDNA5 async global->LDS copy (ASYNCcnt path) -------------------------
// LDS byte offset == low 32 bits of the generic __shared__ pointer
// (LDS aperture maps addr[31:0] directly to the LDS address).
__device__ __forceinline__ uint32_t lds_off(const void* p) {
  return (uint32_t)(uintptr_t)p;
}
__device__ __forceinline__ void async_b128(uint32_t lds_byte_addr, const void* g) {
  asm volatile("global_load_async_to_lds_b128 %0, %1, off"
               :: "v"(lds_byte_addr), "v"((unsigned long long)(uintptr_t)g)
               : "memory");
}
__device__ __forceinline__ void wait_async0() {
  asm volatile("s_wait_asynccnt 0" ::: "memory");
}

// ---------------------------------------------------------------------------
// Batched bf16 WMMA GEMM:  C[z][m,n] = act( sum_k A[z][m,k]*B[z][n,k] + bias[n] )
// Batch offsets: off = (z/nInner)*So + (z%nInner)*Si   (per operand)
// Block: 256 threads = 8 waves; block tile 128x128; wave tile 32x64 (2x4 WMMA).
// K-tiles double-buffered in LDS, filled by global_load_async_to_lds_b128.
// ---------------------------------------------------------------------------
__global__ __launch_bounds__(256) void gemm_bf16_wmma(
    const bfloat* __restrict__ A, int lda, long aSo, long aSi,
    const bfloat* __restrict__ Bm, int ldb, long bSo, long bSi,
    const float* __restrict__ bias,
    float* __restrict__ Cf, bfloat* __restrict__ Cb, int ldc, long cSo, long cSi,
    int K, int nInner, int act)
{
  // 128 rows x 32 K bf16, padded to 48 elems/row so 16B chunks stay aligned
  __shared__ bfloat As[2][128 * 48];
  __shared__ bfloat Bs[2][128 * 48];

  const int z = blockIdx.z;
  const long aOff = (long)(z / nInner) * aSo + (long)(z % nInner) * aSi;
  const long bOff = (long)(z / nInner) * bSo + (long)(z % nInner) * bSi;
  const long cOff = (long)(z / nInner) * cSo + (long)(z % nInner) * cSi;

  const int bm = blockIdx.y * 128;
  const int bn = blockIdx.x * 128;
  const int t    = threadIdx.x;
  const int wave = t >> 5;
  const int lane = t & 31;
  const int half = lane >> 4;
  const int l16  = lane & 15;
  const int wm = (wave & 3) * 32;    // wave row offset in block tile (4 groups)
  const int wn = (wave >> 2) * 64;   // wave col offset in block tile (2 groups)

  v8f acc[8] = {};

  // cooperative tile fill: 256 threads x 32B per tile (2 async b128 each)
  const int ldrow = t >> 1;          // 0..127
  const int ldcol = (t & 1) * 16;    // 0 or 16
  const bfloat* Ag = A + aOff + (long)(bm + ldrow) * lda + ldcol;
  const bfloat* Bg = Bm + bOff + (long)(bn + ldrow) * ldb + ldcol;
  const uint32_t asA = lds_off(&As[0][ldrow * 48 + ldcol]);
  const uint32_t asB = lds_off(&Bs[0][ldrow * 48 + ldcol]);
  const uint32_t bufStride = (uint32_t)(128 * 48 * sizeof(bfloat));   // 12288 B

  const int nk = K >> 5;
  auto issue = [&](int i) {
    const uint32_t sel = (uint32_t)(i & 1) * bufStride;
    const bfloat* ga = Ag + (long)i * 32;
    const bfloat* gb = Bg + (long)i * 32;
    async_b128(asA + sel,      ga);
    async_b128(asA + sel + 16, ga + 8);
    async_b128(asB + sel,      gb);
    async_b128(asB + sel + 16, gb + 8);
  };

  issue(0);
  for (int i = 0; i < nk; ++i) {
    wait_async0();        // my fills for buf[i&1] are done
    __syncthreads();      // -> everyone's fills are done; prior reads retired
    if (i + 1 < nk) issue(i + 1);   // overlap next fill with this compute

    const bfloat* AsB = As[i & 1];
    const bfloat* BsB = Bs[i & 1];
    FragBf fa[2], fb[4];
#pragma unroll
    for (int mi = 0; mi < 2; ++mi) {
      // A 16x32 bf16 layout: half 0 -> K {0..7}U{16..23}; half 1 -> {8..15}U{24..31}
      const bfloat* p = &AsB[(wm + mi * 16 + l16) * 48];
      fa[mi].q[0] = *(const uint4*)(p + half * 8);
      fa[mi].q[1] = *(const uint4*)(p + 16 + half * 8);
    }
#pragma unroll
    for (int ni = 0; ni < 4; ++ni) {
      // B 32x16 bf16 layout: half 0 -> K {0..7}U{16..23}; half 1 -> {8..15}U{24..31}
      const bfloat* p = &BsB[(wn + ni * 16 + l16) * 48];
      fb[ni].q[0] = *(const uint4*)(p + half * 8);
      fb[ni].q[1] = *(const uint4*)(p + 16 + half * 8);
    }
#pragma unroll
    for (int mi = 0; mi < 2; ++mi)
#pragma unroll
      for (int ni = 0; ni < 4; ++ni)
        acc[mi * 4 + ni] = __builtin_amdgcn_wmma_f32_16x16x32_bf16(
            false, fa[mi].v, false, fb[ni].v, (short)0, acc[mi * 4 + ni], false, false);
  }

  // epilogue: C/D layout -> lane l16 = column, M = half*8 + vgpr
#pragma unroll
  for (int mi = 0; mi < 2; ++mi) {
#pragma unroll
    for (int ni = 0; ni < 4; ++ni) {
      const int gm = bm + wm + mi * 16;
      const int gn = bn + wn + ni * 16;
      const int n = gn + l16;
      const float bb = bias ? bias[n] : 0.f;
      v8f a = acc[mi * 4 + ni];
#pragma unroll
      for (int v = 0; v < 8; ++v) {
        const int m = gm + half * 8 + v;
        float val = a[v] + bb;
        if (act == 1)      val = fmaxf(val, 0.f);
        else if (act == 2) val = val / (1.f + __expf(-val));   // SiLU
        const long idx = cOff + (long)m * ldc + n;
        if (Cf) Cf[idx] = val;
        if (Cb) Cb[idx] = (bfloat)val;
      }
    }
  }
}

// ---------------------------------------------------------------------------
// BatchNorm stats: one block per channel, reduce over (B, BINS, W)
// ---------------------------------------------------------------------------
__global__ void bn_stats_kernel(const float* __restrict__ x, float* mean, float* rstd)
{
  const int c = blockIdx.x;
  const long plane = (long)kBINS * kW;
  float s = 0.f, sq = 0.f;
  for (int b = 0; b < kB; ++b) {
    const float* p = x + ((long)b * kC + c) * plane;
    for (long i = threadIdx.x; i < plane; i += blockDim.x) {
      float v = p[i]; s += v; sq += v * v;
    }
  }
  __shared__ float ss[256], sqq[256];
  ss[threadIdx.x] = s; sqq[threadIdx.x] = sq;
  __syncthreads();
  for (int st = 128; st > 0; st >>= 1) {
    if ((int)threadIdx.x < st) { ss[threadIdx.x] += ss[threadIdx.x + st]; sqq[threadIdx.x] += sqq[threadIdx.x + st]; }
    __syncthreads();
  }
  if (threadIdx.x == 0) {
    const float N = (float)(plane * kB);
    const float m = ss[0] / N;
    const float var = sqq[0] / N - m * m;
    mean[c] = m;
    rstd[c] = rsqrtf(var + kEps);
  }
}

// BN affine + channel bottleneck + relu, [B,C,BINS,W] -> [B,W,BINS]
__global__ void bottleneck_kernel(const float* __restrict__ x,
                                  const float* __restrict__ mean, const float* __restrict__ rstd,
                                  const float* __restrict__ g, const float* __restrict__ beta,
                                  const float* __restrict__ bw,
                                  float* __restrict__ out, bfloat* __restrict__ outb)
{
  const long idx = (long)blockIdx.x * blockDim.x + threadIdx.x;
  if (idx >= kOut0) return;
  const int w = (int)(idx % kW);
  const long r = idx / kW;
  const int k = (int)(r % kBINS);
  const int b = (int)(r / kBINS);
  float acc = 0.f;
#pragma unroll
  for (int c = 0; c < kC; ++c) {
    const float xv = x[(((long)b * kC + c) * kBINS + k) * kW + w];
    acc += ((xv - mean[c]) * rstd[c] * g[c] + beta[c]) * bw[c];
  }
  acc = fmaxf(acc, 0.f);
  const long o = ((long)b * kW + w) * kBINS + k;
  out[o] = acc;
  if (outb) outb[o] = (bfloat)acc;
}

// LayerNorm over last dim N; one block (256 thr) per row; f32 and/or bf16 out
__global__ void ln_kernel(const float* __restrict__ in, const float* __restrict__ g,
                          const float* __restrict__ beta,
                          float* __restrict__ out, bfloat* __restrict__ outb, int N)
{
  const long base = (long)blockIdx.x * N;
  float s = 0.f, sq = 0.f;
  for (int i = threadIdx.x; i < N; i += blockDim.x) {
    float v = in[base + i]; s += v; sq += v * v;
  }
  __shared__ float ss[256], sqq[256];
  ss[threadIdx.x] = s; sqq[threadIdx.x] = sq;
  __syncthreads();
  for (int st = 128; st > 0; st >>= 1) {
    if ((int)threadIdx.x < st) { ss[threadIdx.x] += ss[threadIdx.x + st]; sqq[threadIdx.x] += sqq[threadIdx.x + st]; }
    __syncthreads();
  }
  __shared__ float mS, rS;
  if (threadIdx.x == 0) {
    const float m = ss[0] / N;
    const float var = sqq[0] / N - m * m;
    mS = m; rS = rsqrtf(var + kEps);
  }
  __syncthreads();
  const float m = mS, rr = rS;
  for (int i = threadIdx.x; i < N; i += blockDim.x) {
    const float v = (in[base + i] - m) * rr * g[i] + beta[i];
    if (out)  out[base + i] = v;
    if (outb) outb[base + i] = (bfloat)v;
  }
}

// row softmax over N (=512); f32 in, bf16 out
__global__ void softmax_kernel(const float* __restrict__ a, bfloat* __restrict__ out, int N)
{
  const long base = (long)blockIdx.x * N;
  float mx = -3.0e38f;
  for (int i = threadIdx.x; i < N; i += blockDim.x) mx = fmaxf(mx, a[base + i]);
  __shared__ float red[256];
  red[threadIdx.x] = mx; __syncthreads();
  for (int st = 128; st > 0; st >>= 1) {
    if ((int)threadIdx.x < st) red[threadIdx.x] = fmaxf(red[threadIdx.x], red[threadIdx.x + st]);
    __syncthreads();
  }
  __shared__ float mS; if (threadIdx.x == 0) mS = red[0];
  __syncthreads();
  const float m = mS;
  float s = 0.f;
  for (int i = threadIdx.x; i < N; i += blockDim.x) s += __expf(a[base + i] - m);
  __syncthreads();
  red[threadIdx.x] = s; __syncthreads();
  for (int st = 128; st > 0; st >>= 1) {
    if ((int)threadIdx.x < st) red[threadIdx.x] += red[threadIdx.x + st];
    __syncthreads();
  }
  __shared__ float sS; if (threadIdx.x == 0) sS = red[0];
  __syncthreads();
  const float inv = 1.f / sS;
  for (int i = threadIdx.x; i < N; i += blockDim.x)
    out[base + i] = (bfloat)(__expf(a[base + i] - m) * inv);
}

// depthwise conv along W, 'same' pad; in/out layout [B, W, Cch]; out bf16
__global__ void conv_dw_kernel(const float* __restrict__ in, const float* __restrict__ kern,
                               bfloat* __restrict__ out, int Cch, int K)
{
  const long total = (long)kB * kW * Cch;
  const long idx = (long)blockIdx.x * blockDim.x + threadIdx.x;
  if (idx >= total) return;
  const int c = (int)(idx % Cch);
  const long r = idx / Cch;
  const int w = (int)(r % kW);
  const int b = (int)(r / kW);
  const int hk = K / 2;
  float acc = 0.f;
  for (int t = 0; t < K; ++t) {
    const int wi = w + t - hk;
    if (wi >= 0 && wi < kW)
      acc += in[((long)b * kW + wi) * Cch + c] * kern[c * K + t];
  }
  out[idx] = (bfloat)acc;
}

// a = (qk + pT)/sqrt(BINS) + prev   (skew == pure transpose; pT already transposed)
__global__ void score_kernel(const float* __restrict__ qk, const float* __restrict__ pT,
                             const float* __restrict__ prev, float* __restrict__ out, long n)
{
  const long idx = (long)blockIdx.x * blockDim.x + threadIdx.x;
  if (idx < n) out[idx] = (qk[idx] + pT[idx]) * 0.03125f + prev[idx];
}

// out = hL + pad_right(hR)
__global__ void combine_kernel(const float* __restrict__ hL, const float* __restrict__ hR,
                               float* __restrict__ out, int FFw, int halfN)
{
  const long total = kRows * (long)FFw;
  const long idx = (long)blockIdx.x * blockDim.x + threadIdx.x;
  if (idx >= total) return;
  const int f = (int)(idx % FFw);
  const long r = idx / FFw;
  float v = hL[idx];
  if (f < halfN) v += hR[r * halfN + f];
  out[idx] = v;
}

__global__ void add_kernel(const float* __restrict__ a, const float* __restrict__ b,
                           const float* __restrict__ c, float* __restrict__ out, long n)
{
  const long idx = (long)blockIdx.x * blockDim.x + threadIdx.x;
  if (idx < n) out[idx] = a[idx] + b[idx] + (c ? c[idx] : 0.f);
}

__global__ void cast_kernel(const float* __restrict__ in, bfloat* __restrict__ out, long n)
{
  const long idx = (long)blockIdx.x * blockDim.x + threadIdx.x;
  if (idx < n) out[idx] = (bfloat)in[idx];
}

// batched transpose-cast f32 -> bf16: out[z][c,r] = in[z][r,c]
__global__ void transpose_cast_kernel(const float* __restrict__ in, bfloat* __restrict__ out,
                                      int R, int Cc2, int ldin, int ldout,
                                      int nInner, long iSo, long iSi, long oSo, long oSi)
{
  const int z = blockIdx.z;
  const long ioff = (long)(z / nInner) * iSo + (long)(z % nInner) * iSi;
  const long ooff = (long)(z / nInner) * oSo + (long)(z % nInner) * oSi;
  const long idx = (long)blockIdx.x * blockDim.x + threadIdx.x;
  if (idx >= (long)R * Cc2) return;
  const int c = (int)(idx % Cc2);
  const int r = (int)(idx / Cc2);
  out[ooff + (long)c * ldout + r] = (bfloat)in[ioff + (long)r * ldin + c];
}

// d_out[b,0,k,w] = xs[b,w,k]
__global__ void final_out_kernel(const float* __restrict__ xs, float* __restrict__ dout)
{
  const long idx = (long)blockIdx.x * blockDim.x + threadIdx.x;
  if (idx >= kOut0) return;
  const int w = (int)(idx % kW);
  const long r = idx / kW;
  const int k = (int)(r % kBINS);
  const int b = (int)(r / kBINS);
  dout[idx] = xs[((long)b * kW + w) * kBINS + k];
}

// ---------------------------------------------------------------------------
// Host orchestration
// ---------------------------------------------------------------------------
static inline size_t alignup(size_t x) { return (x + 255) & ~(size_t)255; }

extern "C" void kernel_launch(void* const* d_in, const int* in_sizes, int n_in,
                              void* d_out, int out_size, void* d_ws, size_t ws_size,
                              hipStream_t stream)
{
  (void)in_sizes; (void)n_in; (void)out_size; (void)ws_size;

  const float* x     = (const float*)d_in[0];
  const float* mem   = (const float*)d_in[1];
  const float* sa1   = (const float*)d_in[2];
  const float* ea1   = (const float*)d_in[3];
  const float* sa2   = (const float*)d_in[4];
  const float* ea2   = (const float*)d_in[5];
  const float* bn_g  = (const float*)d_in[6];
  const float* bn_b  = (const float*)d_in[7];
  const float* mbn_g = (const float*)d_in[8];
  const float* mbn_b = (const float*)d_in[9];
  const float* bot_w = (const float*)d_in[10];
  const float* mbot_w= (const float*)d_in[11];
  const float* ln_g  = (const float*)d_in[12];
  const float* ln_b  = (const float*)d_in[13];
  const float* ln3_g = (const float*)d_in[14];
  const float* ln3_b = (const float*)d_in[15];
  const float* attn_w= (const float*)d_in[16];
  const float* attn_b= (const float*)d_in[17];
  const float* attn_er=(const float*)d_in[18];
  const float* dw1L  = (const float*)d_in[19];
  const float* pw1L  = (const float*)d_in[20];
  const float* dw1R  = (const float*)d_in[21];
  const float* pw1R  = (const float*)d_in[22];
  const float* dw2   = (const float*)d_in[23];
  const float* pw2   = (const float*)d_in[24];
  const float* w3    = (const float*)d_in[25];
  const float* w4    = (const float*)d_in[26];

  // output regions (f32): out, sa1o, ea1o, sa2o, ea2o
  float* out0    = (float*)d_out;
  float* out_sa1 = out0 + kOut0;
  float* out_ea1 = out_sa1 + kAtt;
  float* out_sa2 = out_ea1 + kAtt;
  float* out_ea2 = out_sa2 + kAtt;

  // workspace allocator
  char* wsb = (char*)d_ws;
  size_t off = 0;
  auto allocF = [&](long n) -> float* { float* p = (float*)(wsb + off); off = alignup(off + (size_t)n * 4); return p; };
  auto allocB = [&](long n) -> bfloat* { bfloat* p = (bfloat*)(wsb + off); off = alignup(off + (size_t)n * 2); return p; };

  const long T  = kRows * kBINS;            // 2M token-feature elems
  const long TF = kRows * kFF;              // 4M

  float* st_mx = allocF(kC);  float* st_rx = allocF(kC);
  float* st_mm = allocF(kC);  float* st_rm = allocF(kC);
  float*  xs   = allocF(T);
  float*  ms   = allocF(T);   bfloat* ms_b  = allocB(T);
  float*  h    = allocF(T);   bfloat* h_b   = allocB(T);
  bfloat* q_b  = allocB(T);
  bfloat* k_b  = allocB(T);
  float*  v_f  = allocF(T);
  bfloat* vT_b = allocB(T);
  bfloat* o_b  = allocB(T);
  float*  pT   = allocF(kAtt);
  float*  qk   = allocF(kAtt);
  bfloat* att_b= allocB(kAtt);
  float*  hs   = allocF(T);
  float*  hm   = allocF(T);
  bfloat* convL= allocB(T);
  bfloat* convR= allocB(T);
  float*  hL   = allocF(TF);
  float*  hR   = allocF(T);
  float*  comb = allocF(TF);
  float*  ln3o = allocF(TF);
  bfloat* conv2= allocB(TF);
  float*  ffo  = allocF(T);
  bfloat* g1_b = allocB(kRows * 2L * kFF);
  float*  w4o  = allocF(T);
  bfloat* aw_b = allocB(16L * kBINS * kBINS);   // all 16 attn matrices
  bfloat* erT_b= allocB(4L * kW * kD);
  bfloat* pw1L_b = allocB((long)kFF * kBINS);
  bfloat* pw1R_b = allocB((long)(kFF / 2) * kBINS);
  bfloat* pw2_b  = allocB((long)kBINS * kFF);
  bfloat* w3_b   = allocB(2L * kFF * kBINS);
  bfloat* w4_b   = allocB((long)kBINS * 2L * kFF);

  auto g1d = [](long n) { return dim3((unsigned)((n + 255) / 256)); };

  // batched GEMM launcher (block tile 128x128, 256 threads)
  auto gemm = [&](const bfloat* A, int lda, long aSo, long aSi,
                  const bfloat* Bm, int ldb, long bSo, long bSi,
                  const float* bias,
                  float* Cf, bfloat* Cb, int ldc, long cSo, long cSi,
                  int M, int N, int K, int batch, int nInner, int act) {
    dim3 grid(N / 128, M / 128, batch);
    gemm_bf16_wmma<<<grid, 256, 0, stream>>>(A, lda, aSo, aSi, Bm, ldb, bSo, bSi, bias,
                                             Cf, Cb, ldc, cSo, cSi, K, nInner, act);
  };

  const long WW = (long)kW * kW;
  const long WB = (long)kW * kBINS;

  // one multiband attention pass (projections + scores; optionally full output)
  auto mba = [&](const bfloat* qsrc, const bfloat* kvsrc, int blk, const float* prev,
                 float* a_out, bool full, float* proj_out) {
    const bfloat* Wq = aw_b + (long)(blk * 4 + 0) * kBINS * kBINS;
    const bfloat* Wk = aw_b + (long)(blk * 4 + 1) * kBINS * kBINS;
    const bfloat* Wv = aw_b + (long)(blk * 4 + 2) * kBINS * kBINS;
    const bfloat* Wo = aw_b + (long)(blk * 4 + 3) * kBINS * kBINS;
    const float* bq = attn_b + (long)(blk * 4 + 0) * kBINS;
    const float* bk = attn_b + (long)(blk * 4 + 1) * kBINS;
    const float* bv = attn_b + (long)(blk * 4 + 2) * kBINS;
    const float* bo = attn_b + (long)(blk * 4 + 3) * kBINS;

    gemm(qsrc, kBINS, 0, 0, Wq, kBINS, 0, 0, bq, nullptr, q_b, kBINS, 0, 0,
         (int)kRows, kBINS, kBINS, 1, 1, 0);
    gemm(kvsrc, kBINS, 0, 0, Wk, kBINS, 0, 0, bk, nullptr, k_b, kBINS, 0, 0,
         (int)kRows, kBINS, kBINS, 1, 1, 0);
    if (full)
      gemm(kvsrc, kBINS, 0, 0, Wv, kBINS, 0, 0, bv, v_f, nullptr, kBINS, 0, 0,
           (int)kRows, kBINS, kBINS, 1, 1, 0);

    // pT[z][w,u] = sum_d er[d,w] * q[z][u,d]   (A = erT shared, B = q slices)
    gemm(erT_b + (long)blk * kW * kD, kD, 0, 0,
         q_b, kBINS, WB, kD, nullptr,
         pT, nullptr, kW, 4 * WW, WW,
         kW, kW, kD, kB * kNB, kNB, 0);
    // qk[z][w,u] = sum_d q[z][w,d] * k[z][u,d]
    gemm(q_b, kBINS, WB, kD,
         k_b, kBINS, WB, kD, nullptr,
         qk, nullptr, kW, 4 * WW, WW,
         kW, kW, kD, kB * kNB, kNB, 0);
    score_kernel<<<g1d(kAtt), 256, 0, stream>>>(qk, pT, prev, a_out, kAtt);

    if (!full) return;
    softmax_kernel<<<dim3((unsigned)(kB * kNB * kW)), 256, 0, stream>>>(a_out, att_b, kW);
    // vT[z][d,w] = v[z][w,d]
    transpose_cast_kernel<<<dim3((unsigned)((WW / 2 + 255) / 256), 1, kB * kNB), 256, 0, stream>>>(
        v_f, vT_b, kW, kD, kBINS, kW, kNB, WB, kD, (long)kNB * kD * kW, (long)kD * kW);
    // o[b,w, n*D+d] = sum_u att[z][w,u] * vT[z][d,u]
    gemm(att_b, kW, 4 * WW, WW,
         vT_b, kW, (long)kNB * kD * kW, (long)kD * kW, nullptr,
         nullptr, o_b, kBINS, WB, kD,
         kW, kD, kW, kB * kNB, kNB, 0);
    // output projection
    gemm(o_b, kBINS, 0, 0, Wo, kBINS, 0, 0, bo, proj_out, nullptr, kBINS, 0, 0,
         (int)kRows, kBINS, kBINS, 1, 1, 0);
  };

  // ---- weight casts ----
  cast_kernel<<<g1d(16L * kBINS * kBINS), 256, 0, stream>>>(attn_w, aw_b, 16L * kBINS * kBINS);
  cast_kernel<<<g1d((long)kFF * kBINS), 256, 0, stream>>>(pw1L, pw1L_b, (long)kFF * kBINS);
  cast_kernel<<<g1d((long)(kFF/2) * kBINS), 256, 0, stream>>>(pw1R, pw1R_b, (long)(kFF/2) * kBINS);
  cast_kernel<<<g1d((long)kBINS * kFF), 256, 0, stream>>>(pw2, pw2_b, (long)kBINS * kFF);
  cast_kernel<<<g1d(2L * kFF * kBINS), 256, 0, stream>>>(w3, w3_b, 2L * kFF * kBINS);
  cast_kernel<<<g1d((long)kBINS * 2L * kFF), 256, 0, stream>>>(w4, w4_b, (long)kBINS * 2L * kFF);
  // erT[i][w,d] = er[i][d,w]
  transpose_cast_kernel<<<dim3((unsigned)(((long)kD * kW + 255) / 256), 1, 4), 256, 0, stream>>>(
      attn_er, erT_b, kD, kW, kW, kD, 1, (long)kD * kW, 0, (long)kW * kD, 0);

  // ---- BN + bottleneck ----
  bn_stats_kernel<<<dim3(kC), 256, 0, stream>>>(x, st_mx, st_rx);
  bn_stats_kernel<<<dim3(kC), 256, 0, stream>>>(mem, st_mm, st_rm);
  bottleneck_kernel<<<g1d(kOut0), 256, 0, stream>>>(x, st_mx, st_rx, bn_g, bn_b, bot_w, xs, nullptr);
  bottleneck_kernel<<<g1d(kOut0), 256, 0, stream>>>(mem, st_mm, st_rm, mbn_g, mbn_b, mbot_w, ms, ms_b);

  // ---- norm1 + self_attn1 + enc_attn1 ----
  ln_kernel<<<dim3((unsigned)kRows), 256, 0, stream>>>(xs, ln_g + 0 * kBINS, ln_b + 0 * kBINS, h, h_b, kBINS);
  mba(h_b, h_b, 0, sa1, out_sa1, true, hs);
  mba(h_b, ms_b, 1, ea1, out_ea1, true, hm);
  add_kernel<<<g1d(T), 256, 0, stream>>>(xs, hs, hm, xs, T);

  // ---- norm2 + conv feed-forward ----
  ln_kernel<<<dim3((unsigned)kRows), 256, 0, stream>>>(xs, ln_g + 1 * kBINS, ln_b + 1 * kBINS, h, nullptr, kBINS);
  conv_dw_kernel<<<g1d(T), 256, 0, stream>>>(h, dw1L, convL, kBINS, 11);
  conv_dw_kernel<<<g1d(T), 256, 0, stream>>>(h, dw1R, convR, kBINS, 7);
  gemm(convL, kBINS, 0, 0, pw1L_b, kBINS, 0, 0, nullptr, hL, nullptr, kFF, 0, 0,
       (int)kRows, kFF, kBINS, 1, 1, 1 /*relu*/);
  gemm(convR, kBINS, 0, 0, pw1R_b, kBINS, 0, 0, nullptr, hR, nullptr, kFF / 2, 0, 0,
       (int)kRows, kFF / 2, kBINS, 1, 1, 0);
  combine_kernel<<<g1d(TF), 256, 0, stream>>>(hL, hR, comb, kFF, kFF / 2);
  ln_kernel<<<dim3((unsigned)kRows), 256, 0, stream>>>(comb, ln3_g, ln3_b, ln3o, nullptr, kFF);
  conv_dw_kernel<<<g1d(TF), 256, 0, stream>>>(ln3o, dw2, conv2, kFF, 7);
  gemm(conv2, kFF, 0, 0, pw2_b, kFF, 0, 0, nullptr, ffo, nullptr, kBINS, 0, 0,
       (int)kRows, kBINS, kFF, 1, 1, 0);
  add_kernel<<<g1d(T), 256, 0, stream>>>(xs, ffo, nullptr, xs, T);

  // ---- norm4 + self_attn2 (scores only; reference discards hs2) ----
  ln_kernel<<<dim3((unsigned)kRows), 256, 0, stream>>>(xs, ln_g + 2 * kBINS, ln_b + 2 * kBINS, h, h_b, kBINS);
  mba(h_b, h_b, 2, sa2, out_sa2, false, nullptr);
  add_kernel<<<g1d(T), 256, 0, stream>>>(xs, h, nullptr, xs, T);

  // ---- norm5 + enc_attn2 (scores only) ----
  ln_kernel<<<dim3((unsigned)kRows), 256, 0, stream>>>(xs, ln_g + 3 * kBINS, ln_b + 3 * kBINS, h, h_b, kBINS);
  mba(h_b, ms_b, 3, ea2, out_ea2, false, nullptr);
  add_kernel<<<g1d(T), 256, 0, stream>>>(xs, h, nullptr, xs, T);

  // ---- norm6 + gated MLP (w3 / SiLU / w4) ----
  ln_kernel<<<dim3((unsigned)kRows), 256, 0, stream>>>(xs, ln_g + 4 * kBINS, ln_b + 4 * kBINS, nullptr, h_b, kBINS);
  gemm(h_b, kBINS, 0, 0, w3_b, kBINS, 0, 0, nullptr, nullptr, g1_b, 2 * kFF, 0, 0,
       (int)kRows, 2 * kFF, kBINS, 1, 1, 2 /*silu*/);
  gemm(g1_b, 2 * kFF, 0, 0, w4_b, 2 * kFF, 0, 0, nullptr, w4o, nullptr, kBINS, 0, 0,
       (int)kRows, kBINS, 2 * kFF, 1, 1, 0);
  add_kernel<<<g1d(T), 256, 0, stream>>>(xs, w4o, nullptr, xs, T);

  // ---- final transpose to [B,1,BINS,W] ----
  final_out_kernel<<<g1d(kOut0), 256, 0, stream>>>(xs, out0);
}
